// Decoder_78237124264042
// MI455X (gfx1250) — matrix operand order repaired and checked by reading, not compiled, not run on hardware
//
#include <hip/hip_runtime.h>

// Problem constants (fixed by the reference)
#define LAY 6
#define NH 8
#define HD 64
#define DMODEL 512
#define DFF_ 1024
#define NB 4
#define TT 1024
#define MROWS (NB * TT) // 4096
#define EPS 1e-3f

typedef __attribute__((ext_vector_type(16))) _Float16 v16h;
typedef __attribute__((ext_vector_type(8)))  float    v8f;
typedef __attribute__((ext_vector_type(4)))  unsigned int u32x4;

union HFrag { v16h h; u32x4 q[2]; };

// ---- WMMA fragment loaders (wave32 lane layouts per CDNA5 ISA 7.12.2) ----
// A 16x32 f16: lane r=lane&15 is row; half hh=lane>>4; elems 0..7 = K[hh*8..+7],
// elems 8..15 = K[16+hh*8..+7].  base is row-major [rows][stride] in LDS.
__device__ inline v16h ldA(const _Float16* base, int stride, int row0, int kb, int lane) {
  int r = lane & 15, hh = lane >> 4;
  const _Float16* p = base + (row0 + r) * stride + kb;
  HFrag f;
  f.q[0] = *(const u32x4*)(p + hh * 8);
  f.q[1] = *(const u32x4*)(p + 16 + hh * 8);
  return f.h;
}
// B 32x16 f16: lane c=lane&15 is column; half hh; elems 0..7 = K[hh*8..+7],
// elems 8..15 = K[16+hh*8..+7];
// base is column-major [cols][stride(K)] in LDS so K is contiguous.
__device__ inline v16h ldB(const _Float16* base, int stride, int col0, int kb, int lane) {
  int c = lane & 15, hh = lane >> 4;
  const _Float16* p = base + (col0 + c) * stride + kb;
  HFrag f;
  f.q[0] = *(const u32x4*)(p + hh * 8);
  f.q[1] = *(const u32x4*)(p + 16 + hh * 8);
  return f.h;
}
#define WMMA_F16(a, b, c) \
  __builtin_amdgcn_wmma_f32_16x16x32_f16(false, (a), false, (b), (short)0, (c), false, false)

// ---------------- embedding + positional encoding ----------------
__global__ __launch_bounds__(256) void embed_kernel(
    const int* __restrict__ x, const float* __restrict__ emb,
    float* __restrict__ h32, _Float16* __restrict__ h16) {
  int row = blockIdx.x;             // n*T + t
  int t = row & (TT - 1);
  int tok = x[row];
  const float sq = 22.62741699796952f; // sqrt(512)
  #pragma unroll
  for (int cc = 0; cc < 2; ++cc) {
    int c = threadIdx.x + cc * 256;
    float e = emb[(size_t)tok * DMODEL + c] * sq;
    int ii = c & ~1;
    // 10000^(-ii/512) = exp(-ln(10000)*ii/512)
    float rate = expf(-9.210340371976184f * (float)ii / (float)DMODEL);
    float ang = (float)t * rate;
    float p = (c & 1) ? cosf(ang) : sinf(ang);
    float v = e + p;
    h32[(size_t)row * DMODEL + c] = v;
    h16[(size_t)row * DMODEL + c] = (_Float16)v;
  }
}

// ---------------- residual add + LayerNorm ----------------
__global__ __launch_bounds__(256) void ln_add_kernel(
    const float* __restrict__ h, const float* __restrict__ add,
    const float* __restrict__ g, const float* __restrict__ b,
    float* __restrict__ outF32, _Float16* __restrict__ outF16) {
  __shared__ float red[256];
  int row = blockIdx.x;
  int tid = threadIdx.x;
  const float* hr = h + (size_t)row * DMODEL;
  const float* ar = add + (size_t)row * DMODEL;
  float v0 = hr[tid] + ar[tid];
  float v1 = hr[tid + 256] + ar[tid + 256];
  red[tid] = v0 + v1; __syncthreads();
  #pragma unroll
  for (int off = 128; off > 0; off >>= 1) {
    if (tid < off) red[tid] += red[tid + off];
    __syncthreads();
  }
  float mean = red[0] * (1.0f / (float)DMODEL);
  __syncthreads();
  float d0 = v0 - mean, d1 = v1 - mean;
  red[tid] = d0 * d0 + d1 * d1; __syncthreads();
  #pragma unroll
  for (int off = 128; off > 0; off >>= 1) {
    if (tid < off) red[tid] += red[tid + off];
    __syncthreads();
  }
  float rstd = rsqrtf(red[0] * (1.0f / (float)DMODEL) + EPS);
  float y0 = g[tid] * d0 * rstd + b[tid];
  float y1 = g[tid + 256] * d1 * rstd + b[tid + 256];
  outF32[(size_t)row * DMODEL + tid] = y0;
  outF32[(size_t)row * DMODEL + tid + 256] = y1;
  outF16[(size_t)row * DMODEL + tid] = (_Float16)y0;
  outF16[(size_t)row * DMODEL + tid + 256] = (_Float16)y1;
}

// ---------------- tiled WMMA GEMM: C = A(f16)[M,K] * B(f32)[K,Nc] + bias ----
// MODE 0: outF32 = C               (FF2)
// MODE 1: outF16 = relu(C)         (FF1)
// MODE 2: scatter C (f16) into q/k/v [N,H,T,D]   (QKV)
template <int MODE>
__global__ __launch_bounds__(256) void gemm_k(
    const _Float16* __restrict__ A, const float* __restrict__ B,
    const float* __restrict__ bias,
    float* __restrict__ outF32, _Float16* __restrict__ outF16,
    _Float16* __restrict__ qv, _Float16* __restrict__ kv, _Float16* __restrict__ vv,
    int Nc, int K) {
  __shared__ _Float16 As[2][128 * 32];  // [row][k], double buffered
  __shared__ _Float16 Bs[2][128 * 32];  // [col][k]  (col-major so K contiguous)
  int tid = threadIdx.x;
  int lane = tid & 31, w = tid >> 5;
  int ln = lane & 15, hh = lane >> 4;
  int waveM = (w >> 1) * 32;  // 4 wave-rows of 32
  int waveN = (w & 1) * 64;   // 2 wave-cols of 64
  int m0 = blockIdx.y * 128, n0 = blockIdx.x * 128;

  v8f acc[2][4];
  #pragma unroll
  for (int i = 0; i < 2; ++i)
    #pragma unroll
    for (int j = 0; j < 4; ++j) acc[i][j] = (v8f)(0.0f);

  int arow = tid >> 1, apart = (tid & 1) * 16;
  int krow = tid >> 3, cpart = (tid & 7) * 16;

  auto stage = [&](int k0, int buf) {
    const _Float16* asrc = A + (size_t)(m0 + arow) * K + k0 + apart;
    *(u32x4*)&As[buf][arow * 32 + apart]     = *(const u32x4*)(asrc);
    *(u32x4*)&As[buf][arow * 32 + apart + 8] = *(const u32x4*)(asrc + 8);
    const float* bsrc = B + (size_t)(k0 + krow) * Nc + n0 + cpart;
    #pragma unroll
    for (int j = 0; j < 16; ++j)
      Bs[buf][(cpart + j) * 32 + krow] = (_Float16)bsrc[j];
  };

  int nk = K >> 5;
  stage(0, 0);
  int cur = 0;
  for (int kt = 0; kt < nk; ++kt) {
    __syncthreads();
    if (kt + 1 < nk) stage((kt + 1) << 5, cur ^ 1);  // overlap next-tile loads with WMMAs
    v16h af[2], bf[4];
    #pragma unroll
    for (int tm = 0; tm < 2; ++tm) af[tm] = ldA(As[cur], 32, waveM + tm * 16, 0, lane);
    #pragma unroll
    for (int tc = 0; tc < 4; ++tc) bf[tc] = ldB(Bs[cur], 32, waveN + tc * 16, 0, lane);
    #pragma unroll
    for (int tm = 0; tm < 2; ++tm)
      #pragma unroll
      for (int tc = 0; tc < 4; ++tc)
        acc[tm][tc] = WMMA_F16(af[tm], bf[tc], acc[tm][tc]);
    cur ^= 1;
  }

  #pragma unroll
  for (int tm = 0; tm < 2; ++tm)
    #pragma unroll
    for (int tc = 0; tc < 4; ++tc)
      #pragma unroll
      for (int e = 0; e < 8; ++e) {
        int row = m0 + waveM + tm * 16 + hh * 8 + e;
        int col = n0 + waveN + tc * 16 + ln;
        float val = acc[tm][tc][e] + bias[col];
        if constexpr (MODE == 0) {
          outF32[(size_t)row * Nc + col] = val;
        } else if constexpr (MODE == 1) {
          outF16[(size_t)row * Nc + col] = (_Float16)fmaxf(val, 0.0f);
        } else {
          // col = head*192 + d*3 + j ; row = n*1024 + t
          int hd = col / 192, rem = col % 192;
          int d = rem / 3, j = rem % 3;
          int n = row >> 10, t = row & (TT - 1);
          size_t dst = (((size_t)(n * NH + hd) * TT) + t) * HD + d;
          _Float16 hv = (_Float16)val;
          if (j == 0) qv[dst] = hv;
          else if (j == 1) kv[dst] = hv;
          else vv[dst] = hv;
        }
      }
}

// ---------------- flash attention (causal), q/k/v f16 [N,H,T,D] ----------------
__global__ __launch_bounds__(256) void attn_kernel(
    const _Float16* __restrict__ q, const _Float16* __restrict__ k,
    const _Float16* __restrict__ v, float* __restrict__ out) {
  __shared__ _Float16 Qs[128 * 64];      // [t_local][d]
  __shared__ _Float16 Ks[64 * 64];       // [s_local][d]  == B-layout for QK^T
  __shared__ _Float16 Vt[64 * 64];       // [d][s_local]  == B-layout for P*V
  __shared__ _Float16 Ps[8 * 16 * 64];   // per-wave P [16][64]

  int nh = blockIdx.x;            // n*H + h
  int n = nh >> 3, hIdx = nh & 7;
  int q0t = blockIdx.y * 128;
  int tid = threadIdx.x;
  int lane = tid & 31, w = tid >> 5;
  int ln = lane & 15, hh = lane >> 4;

  // stage Q block (128 x 64)
  {
    const _Float16* qsrc = q + ((size_t)nh * TT + q0t) * HD;
    int r = tid >> 1, p = (tid & 1) * 32;
    #pragma unroll
    for (int j = 0; j < 4; ++j)
      *(u32x4*)&Qs[r * 64 + p + j * 8] = *(const u32x4*)&qsrc[(size_t)r * 64 + p + j * 8];
  }
  __syncthreads();
  // Q A-fragments are key-block invariant: load once
  v16h aq0 = ldA(Qs, 64, w * 16, 0, lane);
  v16h aq1 = ldA(Qs, 64, w * 16, 32, lane);

  float mprev[8], lrun[8];
  v8f oacc[4];
  #pragma unroll
  for (int e = 0; e < 8; ++e) { mprev[e] = -1e30f; lrun[e] = 0.0f; }
  #pragma unroll
  for (int od = 0; od < 4; ++od) oacc[od] = (v8f)(0.0f);

  int wrow0 = q0t + w * 16;       // first query row owned by this wave
  int nblocks = (q0t >> 6) + 2;   // causal pruning: keys <= q0t+127
  for (int sb = 0; sb < nblocks; ++sb) {
    int s0 = sb * 64;
    __syncthreads();
    // stage K (row-major) and V (transposed)
    {
      int r = tid >> 2, p = (tid & 3) * 16;
      const _Float16* ksrc = k + ((size_t)nh * TT + s0) * HD;
      *(u32x4*)&Ks[r * 64 + p]     = *(const u32x4*)&ksrc[(size_t)r * 64 + p];
      *(u32x4*)&Ks[r * 64 + p + 8] = *(const u32x4*)&ksrc[(size_t)r * 64 + p + 8];
      const _Float16* vsrc = v + ((size_t)nh * TT + s0) * HD;
      #pragma unroll
      for (int j = 0; j < 16; ++j)
        Vt[(p + j) * 64 + r] = vsrc[(size_t)r * 64 + p + j];
    }
    __syncthreads();

    // wave-uniform causal skip: this wave's rows are all above the key block
    if (s0 > wrow0 + 15) continue;
    bool blockMasked = (s0 + 63) > wrow0;  // wave-uniform

    // S = Q K^T  (wave handles 16 query rows x 64 key cols)
    v8f sv[4];
    #pragma unroll
    for (int tc = 0; tc < 4; ++tc) {
      v8f s = (v8f)(0.0f);
      s = WMMA_F16(aq0, ldB(Ks, 64, tc * 16, 0, lane), s);
      s = WMMA_F16(aq1, ldB(Ks, 64, tc * 16, 32, lane), s);
      sv[tc] = s;
    }

    // online softmax in fp32
    const float scale = 0.125f; // 1/sqrt(64)
    #pragma unroll
    for (int e = 0; e < 8; ++e) {
      int abs_t = wrow0 + hh * 8 + e;
      float mx = -3.0e38f;
      #pragma unroll
      for (int tc = 0; tc < 4; ++tc) {
        float val = sv[tc][e] * scale;
        if (blockMasked) {
          int abs_s = s0 + tc * 16 + ln;
          val += (abs_s > abs_t) ? -1e9f : 0.0f;
        }
        sv[tc][e] = val;
        mx = fmaxf(mx, val);
      }
      mx = fmaxf(mx, __shfl_xor(mx, 1));
      mx = fmaxf(mx, __shfl_xor(mx, 2));
      mx = fmaxf(mx, __shfl_xor(mx, 4));
      mx = fmaxf(mx, __shfl_xor(mx, 8));
      float mn = fmaxf(mprev[e], mx);
      float alpha = __expf(mprev[e] - mn);
      mprev[e] = mn;
      float rs = 0.0f;
      #pragma unroll
      for (int tc = 0; tc < 4; ++tc) {
        float pex = __expf(sv[tc][e] - mn);
        sv[tc][e] = pex;
        rs += pex;
      }
      rs += __shfl_xor(rs, 1);
      rs += __shfl_xor(rs, 2);
      rs += __shfl_xor(rs, 4);
      rs += __shfl_xor(rs, 8);
      lrun[e] = lrun[e] * alpha + rs;
      #pragma unroll
      for (int od = 0; od < 4; ++od) oacc[od][e] *= alpha;
    }

    // re-stripe P (C-layout) -> A-fragment layout via per-wave LDS buffer
    _Float16* pw = Ps + w * 1024;
    #pragma unroll
    for (int tc = 0; tc < 4; ++tc)
      #pragma unroll
      for (int e = 0; e < 8; ++e)
        pw[(hh * 8 + e) * 64 + tc * 16 + ln] = (_Float16)sv[tc][e];
    // DS ops are in-order within a wave; wave32 is lockstep -> safe to reload.
    v16h pa0 = ldA(pw, 64, 0, 0, lane);
    v16h pa1 = ldA(pw, 64, 0, 32, lane);
    #pragma unroll
    for (int od = 0; od < 4; ++od) {
      oacc[od] = WMMA_F16(pa0, ldB(Vt, 64, od * 16, 0, lane), oacc[od]);
      oacc[od] = WMMA_F16(pa1, ldB(Vt, 64, od * 16, 32, lane), oacc[od]);
    }
  }

  // normalize and write [N*T, DM] fp32 at columns hIdx*64..+63
  #pragma unroll
  for (int e = 0; e < 8; ++e) {
    float inv = 1.0f / lrun[e];
    int t = q0t + w * 16 + hh * 8 + e;
    size_t rowoff = ((size_t)(n * TT + t)) * DMODEL + hIdx * HD;
    #pragma unroll
    for (int od = 0; od < 4; ++od)
      out[rowoff + od * 16 + ln] = oacc[od][e] * inv;
  }
}

// ---------------- host orchestration ----------------
extern "C" void kernel_launch(void* const* d_in, const int* in_sizes, int n_in,
                              void* d_out, int out_size, void* d_ws, size_t ws_size,
                              hipStream_t stream) {
  (void)in_sizes; (void)n_in; (void)out_size; (void)ws_size;
  const int*   x     = (const int*)d_in[0];
  const float* emb   = (const float*)d_in[1];
  const float* Wqkv  = (const float*)d_in[2];
  const float* bqkv  = (const float*)d_in[3];
  const float* Wff   = (const float*)d_in[4];
  const float* bff   = (const float*)d_in[5];
  const float* Wo    = (const float*)d_in[6];
  const float* bo    = (const float*)d_in[7];
  const float* g1    = (const float*)d_in[8];
  const float* beta1 = (const float*)d_in[9];
  const float* g2    = (const float*)d_in[10];
  const float* beta2 = (const float*)d_in[11];

  char* wp = (char*)d_ws;
  float*    h32    = (float*)wp;    wp += (size_t)MROWS * DMODEL * 4;
  _Float16* h16    = (_Float16*)wp; wp += (size_t)MROWS * DMODEL * 2;
  _Float16* q16    = (_Float16*)wp; wp += (size_t)MROWS * DMODEL * 2;
  _Float16* k16    = (_Float16*)wp; wp += (size_t)MROWS * DMODEL * 2;
  _Float16* v16    = (_Float16*)wp; wp += (size_t)MROWS * DMODEL * 2;
  float*    attn32 = (float*)wp;    wp += (size_t)MROWS * DMODEL * 4;
  _Float16* ff16   = (_Float16*)wp; wp += (size_t)MROWS * DFF_ * 2;
  float*    ff2    = (float*)wp;    wp += (size_t)MROWS * DMODEL * 4;

  embed_kernel<<<MROWS, 256, 0, stream>>>(x, emb, h32, h16);

  for (int l = 0; l < LAY; ++l) {
    const float* Wqkv_l = Wqkv + (size_t)l * DMODEL * 3 * DMODEL;
    const float* bqkv_l = bqkv + (size_t)l * 3 * DMODEL;
    const float* Wff_l  = Wff + (size_t)l * DMODEL * DFF_;
    const float* bff_l  = bff + (size_t)l * DFF_;
    const float* Wo_l   = Wo + (size_t)l * DFF_ * DMODEL;
    const float* bo_l   = bo + (size_t)l * DMODEL;

    // QKV projection with scatter into [N,H,T,D] f16 buffers
    gemm_k<2><<<dim3((3 * DMODEL) / 128, MROWS / 128), 256, 0, stream>>>(
        h16, Wqkv_l, bqkv_l, nullptr, nullptr, q16, k16, v16, 3 * DMODEL, DMODEL);

    // causal flash attention
    attn_kernel<<<dim3(NB * NH, TT / 128), 256, 0, stream>>>(q16, k16, v16, attn32);

    // h = LN1(h + attn)
    ln_add_kernel<<<MROWS, 256, 0, stream>>>(h32, attn32, g1 + (size_t)l * DMODEL,
                                             beta1 + (size_t)l * DMODEL, h32, h16);

    // FF1: relu(h @ Wff + bff) -> f16
    gemm_k<1><<<dim3(DFF_ / 128, MROWS / 128), 256, 0, stream>>>(
        h16, Wff_l, bff_l, nullptr, ff16, nullptr, nullptr, nullptr, DFF_, DMODEL);

    // FF2: ff @ Wo + bo -> fp32
    gemm_k<0><<<dim3(DMODEL / 128, MROWS / 128), 256, 0, stream>>>(
        ff16, Wo_l, bo_l, ff2, nullptr, nullptr, nullptr, nullptr, DMODEL, DFF_);

    // h = LN2(h + ff) ; last layer writes straight to d_out
    float* dst = (l == LAY - 1) ? (float*)d_out : h32;
    ln_add_kernel<<<MROWS, 256, 0, stream>>>(h32, ff2, g2 + (size_t)l * DMODEL,
                                             beta2 + (size_t)l * DMODEL, dst, h16);
  }
}
